// Nnue_24919400251567
// MI455X (gfx1250) — matrix operand order repaired and checked by reading, not compiled, not run on hardware
//
#include <hip/hip_runtime.h>

// NNUE-style eval for MI455X (gfx1250, wave32):
//   k_w1_to_f16 : W1 fp32 -> fp16 staging (128 KB, L2-resident)
//   k_gather    : segment-sum of 32 embedding rows (+bias) -> psqt, crelu(emb) fp16
//   k_mlp       : layer1 via v_wmma_f32_16x16x32_f16 over all 16 experts
//                 (per-row expert select on the D fragment), layers 2/3 VALU.

#define B_SIZE   8192
#define L_FEAT   32
#define ACCUM_D  256
#define NEXPERT  16
#define LEAKF    0.1f
#define CRELU_HI (127.0f/128.0f)

typedef __attribute__((ext_vector_type(16))) _Float16 v16h;
typedef __attribute__((ext_vector_type(8)))  _Float16 v8h;
typedef __attribute__((ext_vector_type(4)))  _Float16 v4h;
typedef __attribute__((ext_vector_type(8)))  float    v8f;

__device__ __forceinline__ float crelu(float x) {
  float c = fminf(fmaxf(x, -1.0f), CRELU_HI);
  return c + LEAKF * (x - c);
}

// ---------------- Kernel 0: W1 fp32 -> fp16 ----------------
__global__ void k_w1_to_f16(const float* __restrict__ W1,
                            _Float16* __restrict__ w1h, int n) {
  int i = blockIdx.x * blockDim.x + threadIdx.x;
  if (i < n) w1h[i] = (_Float16)W1[i];
}

// ---------------- Kernel 1: gather-sum + activation ----------------
// 4 batch elements per 256-thread block; 64 lanes x float4 span the 256 dims.
__global__ __launch_bounds__(256) void k_gather(
    const int* __restrict__ indices, const float* __restrict__ embed,
    const float* __restrict__ bias, _Float16* __restrict__ emb,
    float* __restrict__ psqt) {
  const int sub = threadIdx.x >> 6;      // element within block (0..3)
  const int t   = threadIdx.x & 63;      // 0..63
  const int b   = blockIdx.x * 4 + sub;
  const int d   = t * 4;

  __shared__ int s_idx[4][L_FEAT];
  if (t < L_FEAT) s_idx[sub][t] = indices[b * L_FEAT + t];
  __syncthreads();

  float4 acc = *(const float4*)(bias + d);
  const float* row = embed + (size_t)s_idx[sub][0] * ACCUM_D;
  #pragma unroll 4
  for (int i = 0; i < L_FEAT; ++i) {
    const float* nxt =
        (i + 1 < L_FEAT) ? embed + (size_t)s_idx[sub][i + 1] * ACCUM_D : row;
    __builtin_prefetch(nxt + d, 0, 1);   // global_prefetch_b8 of next row
    float4 v = *(const float4*)(row + d);
    acc.x += v.x; acc.y += v.y; acc.z += v.z; acc.w += v.w;
    row = nxt;
  }
  if (t == 0) psqt[b] = acc.x;           // pre-activation dim 0

  v4h h;
  h[0] = (_Float16)crelu(acc.x);
  h[1] = (_Float16)crelu(acc.y);
  h[2] = (_Float16)crelu(acc.z);
  h[3] = (_Float16)crelu(acc.w);
  *(v4h*)(emb + (size_t)b * ACCUM_D + d) = h;
}

// ---------------- Kernel 2: WMMA layer1 + VALU layers 2/3 ----------------
// One wave per 16-batch tile; 8 waves (8 tiles) per block.
__global__ __launch_bounds__(256) void k_mlp(
    const _Float16* __restrict__ emb, const _Float16* __restrict__ w1h,
    const float* __restrict__ b1, const float* __restrict__ W2,
    const float* __restrict__ b2, const float* __restrict__ W3,
    const float* __restrict__ b3, const int* __restrict__ which_model,
    const int* __restrict__ lengths, const float* __restrict__ psqt,
    float* __restrict__ out) {
  __shared__ float s_h1[8][16][17];      // +1 pad vs bank conflicts

  const int lane = threadIdx.x & 31;
  const int wave = threadIdx.x >> 5;
  const int tile = blockIdx.x * 8 + wave;
  const int m0   = tile * 16;
  const int r    = lane & 15;
  const int half = lane >> 4;

  // A fragments: 16x256 emb tile, 8 K-steps of 32.
  // 16-bit A layout: lane (r, half) holds K in [kb+8*half, +7] and [kb+16+8*half, +7].
  const _Float16* arow = emb + (size_t)(m0 + r) * ACCUM_D;
  v16h afrag[8];
  #pragma unroll
  for (int kk = 0; kk < 8; ++kk) {
    const int kb = kk * 32;
    v8h lo = *(const v8h*)(arow + kb + 8 * half);
    v8h hi = *(const v8h*)(arow + kb + 16 + 8 * half);
    v16h a;
    #pragma unroll
    for (int j = 0; j < 8; ++j) { a[j] = lo[j]; a[j + 8] = hi[j]; }
    afrag[kk] = a;
  }

  // Expert selector for the 8 D-rows this lane owns (M = 8*half + i).
  int   selm[8];
  float selh1[8];
  #pragma unroll
  for (int i = 0; i < 8; ++i) {
    int m   = m0 + 8 * half + i;
    selm[i] = which_model[m] + (lengths[m] / 17) * 4;
    selh1[i] = 0.0f;
  }

  for (int e = 0; e < NEXPERT; ++e) {
    v8f acc = {};
    // B layout: lane's column N = r; K in [kb+8*half, +7] and [kb+16+8*half, +7].
    const _Float16* wrow = w1h + (size_t)(e * 16 + r) * ACCUM_D + 8 * half;
    #pragma unroll
    for (int kk = 0; kk < 8; ++kk) {
      v8h blo = *(const v8h*)(wrow + kk * 32);
      v8h bhi = *(const v8h*)(wrow + kk * 32 + 16);
      v16h bfrag;
      #pragma unroll
      for (int j = 0; j < 8; ++j) { bfrag[j] = blo[j]; bfrag[j + 8] = bhi[j]; }
      acc = __builtin_amdgcn_wmma_f32_16x16x32_f16(
          false, afrag[kk], false, bfrag, (short)0, acc, false, false);
    }
    #pragma unroll
    for (int i = 0; i < 8; ++i)
      if (selm[i] == e) selh1[i] = acc[i];
  }

  // bias + activation, stage selected h1 to LDS (row M = 8*half+i, col N = r).
  #pragma unroll
  for (int i = 0; i < 8; ++i) {
    float x = selh1[i] + b1[selm[i] * 16 + r];
    s_h1[wave][8 * half + i][r] = crelu(x);
  }
  __syncthreads();

  // Layers 2 & 3: lane r handles element m0+r; `half` splits the 32 h2 outputs.
  const int em  = r;
  const int m   = m0 + em;
  const int sel = which_model[m] + (lengths[m] / 17) * 4;
  const float* w2 = W2 + (size_t)sel * 32 * 16;
  const float* w3 = W3 + (size_t)sel * 32;

  float h1loc[16];
  #pragma unroll
  for (int nn = 0; nn < 16; ++nn) h1loc[nn] = s_h1[wave][em][nn];

  float part = 0.0f;
  #pragma unroll
  for (int o = 0; o < 16; ++o) {
    int oo = half * 16 + o;
    float s = b2[sel * 32 + oo];
    #pragma unroll
    for (int nn = 0; nn < 16; ++nn) s += w2[oo * 16 + nn] * h1loc[nn];
    part += w3[oo] * crelu(s);
  }
  part += __shfl_xor(part, 16, 32);       // combine the two output halves (wave32)
  if (half == 0) out[m] = tanhf(part + b3[sel] + psqt[m]);
}

// ---------------- launch ----------------
extern "C" void kernel_launch(void* const* d_in, const int* in_sizes, int n_in,
                              void* d_out, int out_size, void* d_ws, size_t ws_size,
                              hipStream_t stream) {
  const int*   indices     = (const int*)  d_in[0];
  /* d_in[1] offsets: uniform segments of L=32, not needed */
  const int*   which_model = (const int*)  d_in[2];
  const int*   lengths     = (const int*)  d_in[3];
  const float* embed       = (const float*)d_in[4];
  const float* bias        = (const float*)d_in[5];
  const float* W1          = (const float*)d_in[6];
  const float* b1          = (const float*)d_in[7];
  const float* W2          = (const float*)d_in[8];
  const float* b2          = (const float*)d_in[9];
  const float* W3          = (const float*)d_in[10];
  const float* b3          = (const float*)d_in[11];
  float* out = (float*)d_out;

  char* ws = (char*)d_ws;
  _Float16* w1h  = (_Float16*)ws;                              // 128 KiB
  _Float16* embh = (_Float16*)(ws + 131072);                   // 4 MiB
  float*    psqt = (float*)   (ws + 131072 + 4194304);         // 32 KiB

  const int w1n = NEXPERT * 16 * ACCUM_D;
  k_w1_to_f16<<<(w1n + 255) / 256, 256, 0, stream>>>(W1, w1h, w1n);
  k_gather<<<B_SIZE / 4, 256, 0, stream>>>(indices, embed, bias, embh, psqt);
  k_mlp<<<B_SIZE / (16 * 8), 256, 0, stream>>>(embh, w1h, b1, W2, b2, W3, b3,
                                               which_model, lengths, psqt, out);
}
